// InteractionLayer_76897094467620
// MI455X (gfx1250) — compile-verified, matrix operand order checked by reading, not run on hardware
//
#include <hip/hip_runtime.h>
#include <hip/hip_bf16.h>
#include <math.h>

typedef __attribute__((ext_vector_type(16))) _Float16 v16h;
typedef __attribute__((ext_vector_type(8)))  float    v8f;
typedef __attribute__((ext_vector_type(2)))  _Float16 h2;
typedef __attribute__((ext_vector_type(2)))  __fp16   fh2;   // cvt_pkrtz return type

union V16 {             // v16h viewed as 8 packed half2 pairs
    v16h v;
    h2   p[8];
};

static __device__ __forceinline__ h2 pk_f16(float a, float b) {
    fh2 t = __builtin_amdgcn_cvt_pkrtz(a, b);   // v_cvt_pk_f16_f32
    return __builtin_bit_cast(h2, t);
}

#define F_DIM 32
#define A_DIM 32
#define O_DIM 32
#define TILE_M 16

// out = y (residual init); scatter kernel accumulates on top.
__global__ void init_out_kernel(const float* __restrict__ y,
                                float* __restrict__ out, int n) {
    int i = blockIdx.x * blockDim.x + threadIdx.x;
    if (i < n) out[i] = y[i];
}

// Per-wave 16-edge tile:
//   A[e, f*32+a] = ef[e,f] * y[dst[e], a]  (built on the fly, f16)
//   B[f][a, o]   = w[f,a,o]                (pre-swizzled f16 fragments in LDS)
//   C (f32)     += A x B   via v_wmma_f32_16x16x32_f16, 2 N-tiles for O=32,
//   4 independent accumulator chains + 2 alternating A buffers.
__global__ __launch_bounds__(256)
void edge_tp_wmma_kernel(const float* __restrict__ ef,
                         const float* __restrict__ y,
                         const int*   __restrict__ src,
                         const int*   __restrict__ dst,
                         const float* __restrict__ w,
                         const int*   __restrict__ Np,
                         float*       __restrict__ out,
                         int E) {
    // LDS: [f][ntile][lane][16 halves] = exact B-fragment layout, 64 KB
    extern __shared__ _Float16 wfrag[];

    const int tid = threadIdx.x;

    // ---- Stage w -> LDS in B-fragment order, converting f32 -> f16 ----
    // idx = ((f*2 + nt)*32 + lane)*16 + h
    for (int idx = tid; idx < F_DIM * 2 * 32 * 16; idx += blockDim.x) {
        int h   = idx & 15;
        int ln  = (idx >> 4) & 31;
        int fn  = idx >> 9;          // 0..63
        int nt  = fn & 1;
        int f   = fn >> 1;
        int grp = ln >> 4;
        // 16-bit operand K mapping: h 0..7 -> K = h (+16 for h>=8), +8 for upper half-wave
        int a = (h & 7) + ((h >> 3) << 4) + grp * 8;
        int o = (ln & 15) + nt * 16;
        wfrag[idx] = (_Float16)w[(f * A_DIM + a) * O_DIM + o];
    }
    __syncthreads();

    const float scale = (1.0f / 32.0f) / sqrtf((float)Np[0]);  // TP_NORM / N^0.5

    const int lane = tid & 31;
    const int grp  = lane >> 4;    // half-wave group
    const int row  = lane & 15;    // A-matrix row (edge within tile), C column
    const int wave   = blockIdx.x * (blockDim.x >> 5) + (tid >> 5);
    const int nwaves = gridDim.x * (blockDim.x >> 5);
    const int ntiles = (E + TILE_M - 1) / TILE_M;

    for (int tile = wave; tile < ntiles; tile += nwaves) {
        const int e  = tile * TILE_M + row;
        const int ec = (e < E) ? e : (E - 1);   // clamp: keep EXEC all-ones for WMMA

        // y[dst] row in A-fragment half layout: a = grp*8 + {0..7}, 16+grp*8 + {0..7}
        // Packed f32->f16 conversion (v_cvt_pk_f16_f32): every half2 pair is
        // genuinely packed -> A-build lowers to pure v_pk_mul_f16.
        const int d = dst[ec];
        const float4* y4a = (const float4*)(y + (size_t)d * A_DIM + grp * 8);
        const float4* y4b = (const float4*)(y + (size_t)d * A_DIM + 16 + grp * 8);
        float4 p0 = y4a[0], p1 = y4a[1];
        float4 p2 = y4b[0], p3 = y4b[1];
        V16 yd;
        yd.p[0] = pk_f16(p0.x, p0.y);
        yd.p[1] = pk_f16(p0.z, p0.w);
        yd.p[2] = pk_f16(p1.x, p1.y);
        yd.p[3] = pk_f16(p1.z, p1.w);
        yd.p[4] = pk_f16(p2.x, p2.y);
        yd.p[5] = pk_f16(p2.z, p2.w);
        yd.p[6] = pk_f16(p3.x, p3.y);
        yd.p[7] = pk_f16(p3.z, p3.w);

        // edge feature row (32 f32) -> 16 packed half2 (f0..f31)
        const float4* e4 = (const float4*)(ef + (size_t)ec * F_DIM);
        V16 ehlo, ehhi;
        #pragma unroll
        for (int i = 0; i < 4; ++i) {
            float4 t = e4[i];
            ehlo.p[2 * i + 0] = pk_f16(t.x, t.y);
            ehlo.p[2 * i + 1] = pk_f16(t.z, t.w);
        }
        #pragma unroll
        for (int i = 0; i < 4; ++i) {
            float4 t = e4[4 + i];
            ehhi.p[2 * i + 0] = pk_f16(t.x, t.y);
            ehhi.p[2 * i + 1] = pk_f16(t.z, t.w);
        }

        // Opaque per-tile LDS byte offset: blocks LICM/CSE from hoisting the
        // ds_load_b128 B-fragment loads out of the tile loop (avoids the
        // hoist -> 512-VGPR live range -> scratch spill/reload regression).
        unsigned lofs = (unsigned)(lane * 32);
        asm volatile("" : "+v"(lofs));
        const char* wl = (const char*)wfrag + lofs;

        // fragment index = f*2 + nt, each fragment 1024 bytes
        #define LDB(fragidx) (*(const v16h*)(wl + (fragidx) * 1024))

        // 4 accumulator chains: (ntile 0/1) x (even/odd f)
        v8f c0e = {}, c0o = {}, c1e = {}, c1o = {};

        // one pair (4 fragments) of lookahead
        v16h b00 = LDB(0), b01 = LDB(1), b10 = LDB(2), b11 = LDB(3);

        #pragma unroll
        for (int fp = 0; fp < F_DIM / 2; ++fp) {
            // scalars for f0 = 2*fp (lo half) and f1 = 2*fp+1 (hi half)
            h2 sp = (fp < 8) ? ehlo.p[fp] : ehhi.p[fp - 8];
            h2 s0; s0[0] = sp[0]; s0[1] = sp[0];
            h2 s1; s1[0] = sp[1]; s1[1] = sp[1];

            V16 afA, afB;                     // alternating A buffers
            #pragma unroll
            for (int i = 0; i < 8; ++i) afA.p[i] = yd.p[i] * s0;  // v_pk_mul_f16
            #pragma unroll
            for (int i = 0; i < 8; ++i) afB.p[i] = yd.p[i] * s1;

            v16h t00 = b00, t01 = b01, t10 = b10, t11 = b11;
            if (fp < F_DIM / 2 - 1) {
                b00 = LDB(4 * fp + 4); b01 = LDB(4 * fp + 5);
                b10 = LDB(4 * fp + 6); b11 = LDB(4 * fp + 7);
            }

            c0e = __builtin_amdgcn_wmma_f32_16x16x32_f16(
                      false, afA.v, false, t00, (short)0, c0e, false, false);
            c1e = __builtin_amdgcn_wmma_f32_16x16x32_f16(
                      false, afA.v, false, t01, (short)0, c1e, false, false);
            c0o = __builtin_amdgcn_wmma_f32_16x16x32_f16(
                      false, afB.v, false, t10, (short)0, c0o, false, false);
            c1o = __builtin_amdgcn_wmma_f32_16x16x32_f16(
                      false, afB.v, false, t11, (short)0, c1o, false, false);
        }
        #undef LDB

        // C layout: lane grp 0 -> M = r, grp 1 -> M = r + 8; N = lane&15 (+16 for c1)
        #pragma unroll
        for (int r = 0; r < 8; ++r) {
            int m  = r + grp * 8;
            int em = tile * TILE_M + m;
            if (em < E) {
                int sidx = src[em];
                float v0 = (c0e[r] + c0o[r]) * scale;
                float v1 = (c1e[r] + c1o[r]) * scale;
                v0 = v0 / (1.0f + __expf(-v0));   // SiLU
                v1 = v1 / (1.0f + __expf(-v1));
                atomicAdd(&out[(size_t)sidx * O_DIM + (lane & 15)], v0);
                atomicAdd(&out[(size_t)sidx * O_DIM + 16 + (lane & 15)], v1);
            }
        }
    }
}

extern "C" void kernel_launch(void* const* d_in, const int* in_sizes, int n_in,
                              void* d_out, int out_size, void* d_ws, size_t ws_size,
                              hipStream_t stream) {
    const float* ef  = (const float*)d_in[0];
    const float* y   = (const float*)d_in[1];
    const int*   src = (const int*)d_in[2];
    const int*   dst = (const int*)d_in[3];
    const float* w   = (const float*)d_in[4];
    const int*   Np  = (const int*)d_in[5];
    float* out = (float*)d_out;

    const int E      = in_sizes[0] / F_DIM;
    const int nNodes = in_sizes[1] / A_DIM;
    const int nOut   = nNodes * A_DIM;

    init_out_kernel<<<(nOut + 255) / 256, 256, 0, stream>>>(y, out, nOut);

    const int nTiles = (E + TILE_M - 1) / TILE_M;
    int wgs = (nTiles + 7) / 8;          // 8 waves per 256-thread workgroup
    if (wgs > 1024) wgs = 1024;          // grid-stride over remaining tiles
    if (wgs < 1) wgs = 1;
    edge_tp_wmma_kernel<<<wgs, 256, 64 * 1024, stream>>>(ef, y, src, dst, w, Np,
                                                         out, E);
}
